// CinemaScalarImage_59193239274160
// MI455X (gfx1250) — matrix operand order repaired and checked by reading, not compiled, not run on hardware
//
#include <hip/hip_runtime.h>
#include <hip/hip_bf16.h>

typedef _Float16 half8 __attribute__((ext_vector_type(8)));
typedef _Float16 v16h  __attribute__((ext_vector_type(16)));
typedef float    v8f   __attribute__((ext_vector_type(8)));

#define N_POINTS    1048576
#define TSZ         (1 << 19)
#define BLOCK       256
#define PTS_PER_CHUNK 128         // 8 waves * 16 points
#define CHUNKS      4
#define NUM_BLKS    (N_POINTS / (PTS_PER_CHUNK * CHUNKS))   // 2048

#define INV2PI      0.15915494309189535f

// fragment bases (units of 512 halves = one 32x16 f16 B-fragment)
#define F_S1_WIN   0
#define F_S1_WH    4
#define F_S1_WOUT  36
#define F_S2_WIN   38
#define F_S2_WH    46
#define F_S2_WOUT  62
#define NFRAGS     64

// bias offsets (floats)
#define B_S1_BIN   0
#define B_S1_BH    64
#define B_S1_BOUT  320
#define B_S2_BIN   336
#define B_S2_BH    400
#define B_S2_BOUT  528
#define NBIAS      529

#define WS_FRAG_BYTES  (NFRAGS * 512 * 2)      // 65536
#define WS_BIAS_OFF    WS_FRAG_BYTES

#define SMEM_WFRAG_HALVES (NFRAGS * 512)
#define SMEM_STAGE_HALVES (8 * 16 * 64)
#define SMEM_BYTES (SMEM_WFRAG_HALVES*2 + SMEM_STAGE_HALVES*2 + NBIAS*4 + 64)

// ---- WMMA fragment loaders (layouts per CDNA5 ISA 7.12.2, wave32) ----------
// A: 16x32 f16. lanes 0-15: K=0..7 (h0-7), K=16..23 (h8-15); lanes 16-31: K=8..15, K=24..31
__device__ inline v16h load_a_frag(const _Float16* st, int lane, int kt) {
  int base = (lane & 15) * 64 + kt * 32 + ((lane >> 4) & 1) * 8;
  half8 lo = *(const half8*)(st + base);
  half8 hi = *(const half8*)(st + base + 16);
  return __builtin_shufflevector(lo, hi, 0,1,2,3,4,5,6,7,8,9,10,11,12,13,14,15);
}

// B fragments pre-swizzled: 32 lanes x 16 contiguous halves per fragment
__device__ inline v16h load_b_frag(const _Float16* wf, int lane, int frag) {
  const _Float16* p = wf + frag * 512 + lane * 16;
  half8 lo = *(const half8*)(p);
  half8 hi = *(const half8*)(p + 8);
  return __builtin_shufflevector(lo, hi, 0,1,2,3,4,5,6,7,8,9,10,11,12,13,14,15);
}

__device__ inline v8f wmma32(v16h a, v16h b, v8f c) {
  return __builtin_amdgcn_wmma_f32_16x16x32_f16(false, a, false, b, (short)0, c, false, false);
}

// dense layer: Y = sin(w0*(X @ W + b)).
// All WMMAs issue back-to-back (C = inline 0). Activations run v-outer /
// nt-inner so 4 independent fma->v_sin->cvt chains interleave and fill the
// TRANS delay slots. s = w0 * (1/2pi); hw v_sin computes sin(2*pi*x).
template<int KT, int NT, bool SIN>
__device__ inline void dense_layer(_Float16* st, const _Float16* wf, int fragBase,
                                   const float* bias, float s, int lane) {
  asm volatile("s_wait_dscnt 0x0" ::: "memory");
  v16h A[KT];
#pragma unroll
  for (int kt = 0; kt < KT; ++kt) A[kt] = load_a_frag(st, lane, kt);

  v8f acc[NT];
  float bs[NT];
#pragma unroll
  for (int nt = 0; nt < NT; ++nt) {
    bs[nt] = bias[nt * 16 + (lane & 15)] * s;
    v8f c = {0.f,0.f,0.f,0.f,0.f,0.f,0.f,0.f};
#pragma unroll
    for (int kt = 0; kt < KT; ++kt)
      c = wmma32(A[kt], load_b_frag(wf, lane, fragBase + kt * NT + nt), c);
    acc[nt] = c;
  }
  const int m0 = ((lane >> 4) & 1) * 8;
  const int n  = lane & 15;
#pragma unroll
  for (int v = 0; v < 8; ++v) {
#pragma unroll
    for (int nt = 0; nt < NT; ++nt) {
      float x = fmaf(acc[nt][v], s, bs[nt]);
      if (SIN) x = __builtin_amdgcn_sinf(x);   // hw sin(2*pi*x)
      st[(m0 + v) * 64 + nt * 16 + n] = (_Float16)x;
    }
  }
}

// ---- prep kernel: run once, 1 block ----------------------------------------
__device__ inline void prep_wfrag(const float* w, int K, int Ncols, int KT, int NT,
                                  _Float16* dst, int fragBase, int tid) {
  int total = KT * NT * 512;
  for (int idx = tid; idx < total; idx += BLOCK) {
    int frag = idx >> 9, within = idx & 511;
    int lane = within >> 4, h = within & 15;
    int kt = frag / NT, nt = frag - kt * NT;
    int k = kt * 32 + ((h >> 3) << 4) + ((lane >> 4) & 1) * 8 + (h & 7);
    int n = nt * 16 + (lane & 15);
    float v = (k < K && n < Ncols) ? w[k * Ncols + n] : 0.0f;
    dst[(fragBase + frag) * 512 + within] = (_Float16)v;
  }
}

__global__ void __launch_bounds__(BLOCK)
prep_weights(const float* s1_win, const float* s1_bin, const float* s1_wh,
             const float* s1_bh, const float* s1_wout, const float* s1_bout,
             const float* s2_win, const float* s2_bin, const float* s2_wh,
             const float* s2_bh, const float* s2_wout, const float* s2_bout,
             void* ws) {
  _Float16* wf = (_Float16*)ws;
  float* bias  = (float*)((char*)ws + WS_BIAS_OFF);
  const int tid = threadIdx.x;

  prep_wfrag(s1_win, 32, 64, 1, 4, wf, F_S1_WIN, tid);
  for (int i = 0; i < 4; ++i)
    prep_wfrag(s1_wh + i * 4096, 64, 64, 2, 4, wf, F_S1_WH + i * 8, tid);
  prep_wfrag(s1_wout, 64, 16, 2, 1, wf, F_S1_WOUT, tid);
  prep_wfrag(s2_win, 47, 64, 2, 4, wf, F_S2_WIN, tid);
  for (int i = 0; i < 2; ++i)
    prep_wfrag(s2_wh + i * 4096, 64, 64, 2, 4, wf, F_S2_WH + i * 8, tid);
  prep_wfrag(s2_wout, 64, 1, 2, 1, wf, F_S2_WOUT, tid);

  for (int i = tid; i < 64;  i += BLOCK) bias[B_S1_BIN + i] = s1_bin[i];
  for (int i = tid; i < 256; i += BLOCK) bias[B_S1_BH  + i] = s1_bh[i];
  for (int i = tid; i < 16;  i += BLOCK) bias[B_S1_BOUT+ i] = s1_bout[i];
  for (int i = tid; i < 64;  i += BLOCK) bias[B_S2_BIN + i] = s2_bin[i];
  for (int i = tid; i < 128; i += BLOCK) bias[B_S2_BH  + i] = s2_bh[i];
  if (tid == 0) bias[B_S2_BOUT] = s2_bout[0];
}

// ---- main fused kernel -----------------------------------------------------
__global__ void __launch_bounds__(BLOCK)
ngp_siren_fused(const float* __restrict__ pts, const float* __restrict__ table,
                const void* __restrict__ ws, float* __restrict__ out) {
  extern __shared__ __align__(16) char smem_raw[];
  _Float16* wfrag  = (_Float16*)smem_raw;                     // 64 KB
  _Float16* stage  = wfrag + SMEM_WFRAG_HALVES;               // 16 KB
  float*    biases = (float*)(stage + SMEM_STAGE_HALVES);     // ~2.1 KB

  const int tid = threadIdx.x;

  // stream pre-swizzled weight blob into LDS (b128 copies)
  {
    const uint4* src = (const uint4*)ws;
    uint4* dst = (uint4*)wfrag;
#pragma unroll 4
    for (int i = tid; i < WS_FRAG_BYTES / 16; i += BLOCK) dst[i] = src[i];
    const float* bs = (const float*)((const char*)ws + WS_BIAS_OFF);
    for (int i = tid; i < NBIAS; i += BLOCK) biases[i] = bs[i];
  }
  __syncthreads();

  const int wid  = tid >> 5;
  const int lane = tid & 31;
  const int grp  = lane >> 4;
  const int p    = lane & 15;
  _Float16* st = stage + wid * (16 * 64);
  const float4* tb4 = (const float4*)table;

  for (int chunk = 0; chunk < CHUNKS; ++chunk) {
    const int gpt0 = (blockIdx.x * CHUNKS + chunk) * PTS_PER_CHUNK + wid * 16;
    const int gpt  = gpt0 + p;

    float px = (pts[gpt * 3 + 0] + 1.0f) * 0.5f;
    float py = (pts[gpt * 3 + 1] + 1.0f) * 0.5f;
    float pz = (pts[gpt * 3 + 2] + 1.0f) * 0.5f;

    // ---- hash-grid encoding: this lane does 4 levels x 8 corners ----
    v16h encV;
#pragma unroll
    for (int j = 0; j < 4; ++j) {
      const float RL[4] = {16.f, 35.f, 78.f, 172.f};
      const float RH[4] = {380.f, 840.f, 1854.f, 4096.f};
      float r = grp ? RH[j] : RL[j];
      int   l = grp * 4 + j;
      float sx = px * r, sy = py * r, sz = pz * r;
      float bx = floorf(sx), by = floorf(sy), bz = floorf(sz);
      float fx = sx - bx, fy = sy - by, fz = sz - bz;
      uint32_t ix = (uint32_t)(int)bx, iy = (uint32_t)(int)by, iz = (uint32_t)(int)bz;
      float f0 = 0.f, f1 = 0.f, f2 = 0.f, f3 = 0.f;
#pragma unroll
      for (int c = 0; c < 8; ++c) {
        uint32_t ox = (c >> 2) & 1, oy = (c >> 1) & 1, oz = c & 1;
        uint32_t h = (ix + ox) ^ ((iy + oy) * 2654435761u) ^ ((iz + oz) * 805459861u);
        uint32_t idx = h & (uint32_t)(TSZ - 1);
        float4 g = tb4[(size_t)l * TSZ + idx];
        float w = (ox ? fx : 1.f - fx) * (oy ? fy : 1.f - fy) * (oz ? fz : 1.f - fz);
        f0 += w * g.x; f1 += w * g.y; f2 += w * g.z; f3 += w * g.w;
      }
      encV[j * 4 + 0] = (_Float16)f0;
      encV[j * 4 + 1] = (_Float16)f1;
      encV[j * 4 + 2] = (_Float16)f2;
      encV[j * 4 + 3] = (_Float16)f3;
    }
    {
      half8 e0 = __builtin_shufflevector(encV, encV, 0,1,2,3,4,5,6,7);
      half8 e1 = __builtin_shufflevector(encV, encV, 8,9,10,11,12,13,14,15);
      *(half8*)(st + p * 64 + grp * 16)     = e0;
      *(half8*)(st + p * 64 + grp * 16 + 8) = e1;
    }

    // ---- SIREN 1 ----
    dense_layer<1, 4, true>(st, wfrag, F_S1_WIN, biases + B_S1_BIN, 30.0f * INV2PI, lane);
#pragma unroll
    for (int i = 0; i < 4; ++i)
      dense_layer<2, 4, true>(st, wfrag, F_S1_WH + i * 8, biases + B_S1_BH + i * 64, INV2PI, lane);

    // output layer (linear): density (col 0, relu) / scalar (cols 1..15 -> stage)
    {
      asm volatile("s_wait_dscnt 0x0" ::: "memory");
      v16h A0 = load_a_frag(st, lane, 0);
      v16h A1 = load_a_frag(st, lane, 1);
      int n = lane & 15;
      float b = biases[B_S1_BOUT + n];
      v8f acc = {0.f,0.f,0.f,0.f,0.f,0.f,0.f,0.f};
      acc = wmma32(A0, load_b_frag(wfrag, lane, F_S1_WOUT + 0), acc);
      acc = wmma32(A1, load_b_frag(wfrag, lane, F_S1_WOUT + 1), acc);
#pragma unroll
      for (int v = 0; v < 8; ++v) {
        float x = acc[v] + b;
        int m = grp * 8 + v;
        if (n == 0) out[N_POINTS + gpt0 + m] = fmaxf(x, 0.0f);
        else        st[m * 64 + (n - 1)] = (_Float16)x;
      }
    }
    // append encoding at features 15..46, zero-pad 47..63
#pragma unroll
    for (int t = 0; t < 16; ++t)
      st[p * 64 + 15 + grp * 16 + t] = encV[t];
    if (grp == 1) {
#pragma unroll
      for (int t = 47; t < 64; ++t) st[p * 64 + t] = (_Float16)0.0f;
    }

    // ---- SIREN 2 ----
    dense_layer<2, 4, true>(st, wfrag, F_S2_WIN, biases + B_S2_BIN, 30.0f * INV2PI, lane);
#pragma unroll
    for (int i = 0; i < 2; ++i)
      dense_layer<2, 4, true>(st, wfrag, F_S2_WH + i * 8, biases + B_S2_BH + i * 64, INV2PI, lane);

    {
      asm volatile("s_wait_dscnt 0x0" ::: "memory");
      v16h A0 = load_a_frag(st, lane, 0);
      v16h A1 = load_a_frag(st, lane, 1);
      float b = biases[B_S2_BOUT];
      v8f acc = {0.f,0.f,0.f,0.f,0.f,0.f,0.f,0.f};
      acc = wmma32(A0, load_b_frag(wfrag, lane, F_S2_WOUT + 0), acc);
      acc = wmma32(A1, load_b_frag(wfrag, lane, F_S2_WOUT + 1), acc);
      if ((lane & 15) == 0) {
#pragma unroll
        for (int v = 0; v < 8; ++v)
          out[gpt0 + grp * 8 + v] = acc[v] + b;
      }
    }
  }
}

extern "C" void kernel_launch(void* const* d_in, const int* in_sizes, int n_in,
                              void* d_out, int out_size, void* d_ws, size_t ws_size,
                              hipStream_t stream) {
  (void)in_sizes; (void)n_in; (void)ws_size; (void)out_size;
  const float* pts     = (const float*)d_in[0];
  const float* table   = (const float*)d_in[1];

  hipLaunchKernelGGL(prep_weights, dim3(1), dim3(BLOCK), 0, stream,
                     (const float*)d_in[2], (const float*)d_in[3],
                     (const float*)d_in[4], (const float*)d_in[5],
                     (const float*)d_in[6], (const float*)d_in[7],
                     (const float*)d_in[8], (const float*)d_in[9],
                     (const float*)d_in[10], (const float*)d_in[11],
                     (const float*)d_in[12], (const float*)d_in[13], d_ws);

  hipLaunchKernelGGL(ngp_siren_fused, dim3(NUM_BLKS), dim3(BLOCK), SMEM_BYTES, stream,
                     pts, table, (const void*)d_ws, (float*)d_out);
}
